// Attention_8993661518662
// MI455X (gfx1250) — matrix operand order repaired and checked by reading, not compiled, not run on hardware
//
#include <hip/hip_runtime.h>

typedef __attribute__((ext_vector_type(16))) _Float16 v16h;
typedef __attribute__((ext_vector_type(8)))  _Float16 v8h;
typedef __attribute__((ext_vector_type(4)))  _Float16 v4h;
typedef __attribute__((ext_vector_type(2)))  _Float16 v2h;
typedef __attribute__((ext_vector_type(8)))  float    v8f;

#define QL   2048
#define KVL  2048
#define NH   16
#define HD   128
#define ROWS (NH * HD)      // 2048 floats per token row

#define KSH 136             // K tile LDS row stride (halves): 272B rows, 16B aligned, conflict-free
#define VSH 40              // V^T tile LDS row stride (halves): 80B rows, 16B aligned
#define PSH 34              // P scratch row stride (halves), v2h-aligned

#define LOG2E  1.4426950408889634f
#define NEGBIG (-1e30f)

// ds_swizzle group-of-32 XOR patterns: offset = and(0x1f) | xor<<10
__device__ __forceinline__ float swz1(float v) { return __int_as_float(__builtin_amdgcn_ds_swizzle(__float_as_int(v), 0x1f | (1 << 10))); }
__device__ __forceinline__ float swz2(float v) { return __int_as_float(__builtin_amdgcn_ds_swizzle(__float_as_int(v), 0x1f | (2 << 10))); }
__device__ __forceinline__ float swz4(float v) { return __int_as_float(__builtin_amdgcn_ds_swizzle(__float_as_int(v), 0x1f | (4 << 10))); }
__device__ __forceinline__ float swz8(float v) { return __int_as_float(__builtin_amdgcn_ds_swizzle(__float_as_int(v), 0x1f | (8 << 10))); }

__device__ __forceinline__ float redsum16(float v) {
  v += swz1(v); v += swz2(v); v += swz4(v); v += swz8(v);
  return v;
}

// Load two contiguous 8-half vectors and pair them into a 16-half fragment.
__device__ __forceinline__ v16h load_frag16(const _Float16* p) {
  v8h lo = *(const v8h*)p;
  v8h hi = *(const v8h*)(p + 8);
  return __builtin_shufflevector(lo, hi, 0, 1, 2, 3, 4, 5, 6, 7,
                                         8, 9, 10, 11, 12, 13, 14, 15);
}

__launch_bounds__(256, 1)
__global__ void alibi_flash_attn_wmma(const float* __restrict__ Q,
                                      const float* __restrict__ K,
                                      const float* __restrict__ V,
                                      const float* __restrict__ AB,   // [H, KVL] alibi biases
                                      const float* __restrict__ smp,  // sm_scale scalar
                                      const float* __restrict__ capp, // soft cap scalar
                                      float* __restrict__ Out) {
  __shared__ _Float16 Kh [2][32 * KSH];   // K tile f16, row-major [kv][d], ping-pong
  __shared__ _Float16 Vth[2][HD * VSH];   // V tile f16, transposed [d][kv], ping-pong
  __shared__ _Float16 Ph [8 * 16 * PSH];  // per-wave P scratch (C->A transpose)

  const int tid  = threadIdx.x;
  const int wv   = tid >> 5;
  const int lane = tid & 31;
  const int n    = lane & 15;             // N / column index within 16-group
  const int hi   = lane >> 4;             // half-group selector

  const int head = blockIdx.y;
  const int qb0  = blockIdx.x * 128;      // 8 waves x 16 rows
  const int q0   = qb0 + wv * 16;         // this wave's first q row
  const int q0u  = __builtin_amdgcn_readfirstlane(q0);  // wave-uniform scalar branch

  const float sm_scale = smp[0];
  const float cap      = capp[0];
  const float twocap   = 2.f * cap;
  // t = exp2(c1 * S) == e^(2*S/cap);  score+bias-cap = bias - 2cap/(1+t)
  const float c1 = 2.f * LOG2E * __builtin_amdgcn_rcpf(cap);

  // ---- Q tile -> A fragments (16x32 f16, contiguous k = 16*hi + s), sm_scale folded in ----
  // For j in a 4-group, the d-pairs {0,2,4,6}+base are 8 contiguous floats:
  // load each (kk, half) chunk as two b128s instead of four b64s.
  v16h Qa[4];
  {
    const float* qrow = Q + (size_t)(q0 + n) * ROWS + head * HD;
    #pragma unroll
    for (int kk = 0; kk < 4; ++kk) {
      #pragma unroll
      for (int hb = 0; hb < 2; ++hb) {    // hb=0 -> j 0..3, hb=1 -> j 4..7
        const float* p = qrow + 32 * kk + 16 * hi + 8 * hb;
        float4 a = *(const float4*)p;
        float4 b = *(const float4*)(p + 4);
        int j0 = 4 * hb;
        Qa[kk][2 * (j0 + 0)]     = (_Float16)(a.x * sm_scale);
        Qa[kk][2 * (j0 + 0) + 1] = (_Float16)(a.y * sm_scale);
        Qa[kk][2 * (j0 + 1)]     = (_Float16)(a.z * sm_scale);
        Qa[kk][2 * (j0 + 1) + 1] = (_Float16)(a.w * sm_scale);
        Qa[kk][2 * (j0 + 2)]     = (_Float16)(b.x * sm_scale);
        Qa[kk][2 * (j0 + 2) + 1] = (_Float16)(b.y * sm_scale);
        Qa[kk][2 * (j0 + 3)]     = (_Float16)(b.z * sm_scale);
        Qa[kk][2 * (j0 + 3) + 1] = (_Float16)(b.w * sm_scale);
      }
    }
  }

  v8f Oacc[8];
  #pragma unroll
  for (int dt = 0; dt < 8; ++dt)
    #pragma unroll
    for (int e = 0; e < 8; ++e) Oacc[dt][e] = 0.f;

  // Fixed-shift softmax: exponent shift = cap (scores are capped to +/-cap and
  // bias <= 0, so p = exp(s + bias - cap) can never overflow; underflowing
  // terms are <= e^-60 relative to the row max -- below fp32 softmax noise).
  float psum[8];
  #pragma unroll
  for (int j = 0; j < 8; ++j) psum[j] = 0.f;

  _Float16* pbase = &Ph[wv * 16 * PSH];
  const int kv_end = qb0 + 128;           // causal upper bound for this block

  // Per-thread staging registers (software pipeline): 16 floats of K, 16 of V.
  float4 kreg[4], vreg[4];

  auto load_tile = [&](int kv0) {
    #pragma unroll
    for (int i = 0; i < 4; ++i) {
      int c  = tid + 256 * i;             // 1024 float4 chunks per tensor
      int r  = c >> 5;
      int dc = (c & 31) << 2;
      const float* kp = K + (size_t)(kv0 + r) * ROWS + head * HD + dc;
      const float* vp = V + (size_t)(kv0 + r) * ROWS + head * HD + dc;
      kreg[i] = *(const float4*)kp;
      vreg[i] = *(const float4*)vp;
      if (kv0 + 32 < kv_end) {
        __builtin_prefetch(kp + 32 * ROWS, 0, 1);   // global_prefetch_b8 (depth-2)
        __builtin_prefetch(vp + 32 * ROWS, 0, 1);
      }
    }
  };

  auto store_tile = [&](int buf) {        // cvt once per block, f16 into LDS
    #pragma unroll
    for (int i = 0; i < 4; ++i) {
      int c  = tid + 256 * i;
      int r  = c >> 5;
      int dc = (c & 31) << 2;
      v4h kh;
      kh[0] = (_Float16)kreg[i].x; kh[1] = (_Float16)kreg[i].y;
      kh[2] = (_Float16)kreg[i].z; kh[3] = (_Float16)kreg[i].w;
      *(v4h*)&Kh[buf][r * KSH + dc] = kh;                 // ds_store_b64
      Vth[buf][(dc + 0) * VSH + r] = (_Float16)vreg[i].x; // transposed scatter
      Vth[buf][(dc + 1) * VSH + r] = (_Float16)vreg[i].y;
      Vth[buf][(dc + 2) * VSH + r] = (_Float16)vreg[i].z;
      Vth[buf][(dc + 3) * VSH + r] = (_Float16)vreg[i].w;
    }
  };

  load_tile(0);                           // pipeline prologue

  for (int kv0 = 0; kv0 < kv_end; kv0 += 32) {
    const int buf = (kv0 >> 5) & 1;
    store_tile(buf);                      // commit tile t (regs -> f16 LDS)
    __syncthreads();                      // tile t visible; compute(t-2) done before overwrite
    if (kv0 + 32 < kv_end) load_tile(kv0 + 32);  // issue HBM loads for t+1 early

    if (kv0 > q0u + 15) continue;         // scalar branch: EXEC stays all-ones

    // ---- S = (Q*scale) @ K^T : two 16-wide column tiles, K-dim 128 ----
    v8f S0 = {0.f,0.f,0.f,0.f,0.f,0.f,0.f,0.f};
    v8f S1 = {0.f,0.f,0.f,0.f,0.f,0.f,0.f,0.f};
    #pragma unroll
    for (int kk = 0; kk < 4; ++kk) {
      v16h kb0 = load_frag16(&Kh[buf][n        * KSH + 32 * kk + 16 * hi]);
      v16h kb1 = load_frag16(&Kh[buf][(n + 16) * KSH + 32 * kk + 16 * hi]);
      S0 = __builtin_amdgcn_wmma_f32_16x16x32_f16(false, Qa[kk], false, kb0, (short)0, S0, false, false);
      S1 = __builtin_amdgcn_wmma_f32_16x16x32_f16(false, Qa[kk], false, kb1, (short)0, S1, false, false);
    }

    // ---- soft cap + alibi bias (sm = score + bias - cap, fused tanh) ----
    const float b0 = AB[head * KVL + kv0 + n];
    const float b1 = AB[head * KVL + kv0 + 16 + n];

    float sm0[8], sm1[8];
    #pragma unroll
    for (int j = 0; j < 8; ++j) {
      float t0 = __builtin_amdgcn_exp2f(S0[j] * c1);
      float t1 = __builtin_amdgcn_exp2f(S1[j] * c1);
      sm0[j] = b0 - twocap * __builtin_amdgcn_rcpf(1.f + t0);
      sm1[j] = b1 - twocap * __builtin_amdgcn_rcpf(1.f + t1);
    }

    // ---- causal mask: only the (rare) diagonal tile, scalar-branched ----
    if (__builtin_expect(kv0 + 31 > q0u, 0)) {
      #pragma unroll
      for (int j = 0; j < 8; ++j) {
        int q_row = q0 + j + 8 * hi;
        if (kv0 + n      > q_row) sm0[j] = NEGBIG;
        if (kv0 + 16 + n > q_row) sm1[j] = NEGBIG;
      }
    }

    // ---- fixed-shift exp + per-lane partial row sums ----
    float p0[8], p1[8];
    #pragma unroll
    for (int j = 0; j < 8; ++j) {
      float e0 = __builtin_amdgcn_exp2f(sm0[j] * LOG2E);
      float e1 = __builtin_amdgcn_exp2f(sm1[j] * LOG2E);
      psum[j] += e0 + e1;
      p0[j] = e0; p1[j] = e1;
    }

    // ---- P: C-layout -> A-layout via per-wave f16 LDS (DS in-order per wave) ----
    #pragma unroll
    for (int j = 0; j < 8; ++j) {
      int mr = j + 8 * hi;
      pbase[mr * PSH + n]      = (_Float16)p0[j];
      pbase[mr * PSH + 16 + n] = (_Float16)p1[j];
    }
    v16h Pa;
    #pragma unroll
    for (int j = 0; j < 8; ++j) {
      int k = 16 * hi + 2 * j;
      v2h f = *(const v2h*)&pbase[n * PSH + k];
      Pa[2 * j]     = f[0];
      Pa[2 * j + 1] = f[1];
    }

    // ---- O += P @ V : B fragments straight from transposed f16 V tile ----
    #pragma unroll
    for (int dt = 0; dt < 8; ++dt) {
      v16h vb = load_frag16(&Vth[buf][(16 * dt + n) * VSH + 16 * hi]);
      Oacc[dt] = __builtin_amdgcn_wmma_f32_16x16x32_f16(false, Pa, false, vb, (short)0, Oacc[dt], false, false);
    }
  }

  // ---- single end-of-kernel row-sum reduction, normalize, write [q,h,d] ----
  float rinv[8];
  #pragma unroll
  for (int j = 0; j < 8; ++j) rinv[j] = __builtin_amdgcn_rcpf(redsum16(psum[j]));
  #pragma unroll
  for (int dt = 0; dt < 8; ++dt) {
    #pragma unroll
    for (int j = 0; j < 8; ++j) {
      int qr = q0 + j + 8 * hi;
      Out[(size_t)qr * ROWS + head * HD + 16 * dt + n] = Oacc[dt][j] * rinv[j];
    }
  }
}

extern "C" void kernel_launch(void* const* d_in, const int* in_sizes, int n_in,
                              void* d_out, int out_size, void* d_ws, size_t ws_size,
                              hipStream_t stream) {
  (void)in_sizes; (void)n_in; (void)out_size; (void)d_ws; (void)ws_size;
  const float* q   = (const float*)d_in[0];
  const float* k   = (const float*)d_in[1];
  const float* v   = (const float*)d_in[2];
  const float* ab  = (const float*)d_in[3];
  // d_in[4] = mask (causal tril) -- applied analytically in-kernel
  const float* sms = (const float*)d_in[5];
  const float* cap = (const float*)d_in[6];
  float* out = (float*)d_out;

  dim3 grid(QL / 128, NH);
  dim3 block(256);
  alibi_flash_attn_wmma<<<grid, block, 0, stream>>>(q, k, v, ab, sms, cap, out);
}
